// AttentionSequencePoolingLayer_55791625175488
// MI455X (gfx1250) — matrix operand order repaired and checked by reading, not compiled, not run on hardware
//
#include <hip/hip_runtime.h>
#include <hip/hip_bf16.h>
#include <math.h>

// ---------------------------------------------------------------------------
// AttentionSequencePoolingLayer, fused for MI455X (gfx1250, wave32, WMMA).
//   B=4096, T=200, E=64, H1=80, H2=40
// One workgroup of 256 threads (8 wave32) per batch row.
//   GEMM1: keys[208,64](f16) x Weff[64,80](f16) -> sigmoid -> h1 (WMMA f16->f32)
//   GEMM2: h1[208,96p](f16) x W2[96p,48p](f16)  -> sigmoid -> h2 (WMMA)
//   GEMM3: h2[208,40] . W3 -> scores, mask, softmax, weighted key sum (VALU)
// Sigmoid/softmax use v_rcp_f32 (approx) instead of IEEE div sequences.
// ---------------------------------------------------------------------------

typedef _Float16 v16h __attribute__((ext_vector_type(16)));
typedef _Float16 h8   __attribute__((ext_vector_type(8)));
typedef float    v8f  __attribute__((ext_vector_type(8)));

#define T_DIM      200
#define T_PAD      208          // 13 tiles of 16
#define E_DIM      64
#define H1_DIM     80           // 5 tiles of 16
#define H1_PAD     96           // 3 k-steps of 32
#define H2_DIM     40
#define H2_PAD     48           // 3 tiles of 16
#define NEG_PAD_F  (-4294967295.0f)

// LDS byte offsets (all 32B aligned)
#define OFF_KEYSA  0            // _Float16[208*64]   26624
#define OFF_H1     26624        // _Float16[208*96]   39936
#define OFF_WEFF   66560        // _Float16[80*64]    10240 (col-major [n][k])
#define OFF_W2S    76800        // _Float16[48*96]     9216 (col-major [n][k], zero pad)
#define OFF_H2     86016        // _Float16[208*40]   16640
#define OFF_QS     102656       // float[64]
#define OFF_CVEC   102912       // float[80]
#define OFF_B2S    103232       // float[48]
#define OFF_W3S    103424       // float[40] (+pad)
#define OFF_SC     103584       // float[208]
#define OFF_RED    104416       // float[256]
#define OFF_OUTA   105440       // float[256]
#define SMEM_BYTES 106464

__device__ __forceinline__ v16h load_frag(const _Float16* b0, const _Float16* b1) {
  h8 lo = *(const h8*)b0;
  h8 hi = *(const h8*)b1;
  v16h r;
#pragma unroll
  for (int i = 0; i < 8; ++i) { r[i] = lo[i]; r[8 + i] = hi[i]; }
  return r;
}

// sigmoid via v_exp_f32 + v_rcp_f32 (no IEEE div refinement chain)
__device__ __forceinline__ float sigmoidf_fast(float x) {
  return __builtin_amdgcn_rcpf(1.0f + __expf(-x));
}

__global__ void attn_seq_pool_kernel(const float* __restrict__ queries,   // [B,64]
                                     const float* __restrict__ keys,      // [B,200,64]
                                     const unsigned char* __restrict__ key_masks, // [B,1,200] bool
                                     const float* __restrict__ W1,        // [256,80]
                                     const float* __restrict__ b1,        // [80]
                                     const float* __restrict__ W2,        // [80,40]
                                     const float* __restrict__ b2,        // [40]
                                     const float* __restrict__ W3,        // [40]
                                     const float* __restrict__ b3,        // [1]
                                     float* __restrict__ out)             // [B,1,64]
{
  extern __shared__ char smem[];
  _Float16* keysA = (_Float16*)(smem + OFF_KEYSA);
  _Float16* h1    = (_Float16*)(smem + OFF_H1);
  _Float16* Weff  = (_Float16*)(smem + OFF_WEFF);
  _Float16* W2s   = (_Float16*)(smem + OFF_W2S);
  _Float16* h2    = (_Float16*)(smem + OFF_H2);
  float* qs     = (float*)(smem + OFF_QS);
  float* cvec   = (float*)(smem + OFF_CVEC);
  float* b2s    = (float*)(smem + OFF_B2S);
  float* w3s    = (float*)(smem + OFF_W3S);
  float* scores = (float*)(smem + OFF_SC);
  float* red    = (float*)(smem + OFF_RED);
  float* outacc = (float*)(smem + OFF_OUTA);

  const int tid  = threadIdx.x;
  const int wave = tid >> 5;
  const int lane = tid & 31;
  const int b    = blockIdx.x;

  const float* keysb = keys + (size_t)b * T_DIM * E_DIM;

  // ---------------- Phase 0: stage constants + keys into LDS ----------------
  if (tid < 64) qs[tid] = queries[(size_t)b * E_DIM + tid];
  if (tid < 48) b2s[tid] = (tid < H2_DIM) ? b2[tid] : 0.0f;
  if (tid >= 64 && tid < 64 + H2_DIM) w3s[tid - 64] = W3[tid - 64];

  for (int i = tid; i < T_PAD * E_DIM; i += 256) {
    int r = i >> 6, c = i & 63;
    keysA[i] = (_Float16)((r < T_DIM) ? keysb[r * E_DIM + c] : 0.0f);
  }
  // zero-pad h1 columns 80..95 (K padding for GEMM2)
  for (int i = tid; i < T_PAD * (H1_PAD - H1_DIM); i += 256) {
    int r = i >> 4, c = H1_DIM + (i & 15);
    h1[r * H1_PAD + c] = (_Float16)0.0f;
  }
  // W2 column-major, zero-padded to [48][96]
  for (int i = tid; i < H2_PAD * H1_PAD; i += 256) {
    int n = i / H1_PAD, k = i - n * H1_PAD;
    float w = (n < H2_DIM && k < H1_DIM) ? W2[k * H2_DIM + n] : 0.0f;
    W2s[i] = (_Float16)w;
  }
  __syncthreads();

  // Per-batch effective GEMM1 weight:
  //   Weff[:,n] = W1b - W1c + q (*) W1d     (column-major [n][k] in LDS)
  //   cvec[n]   = q . (W1a + W1c)[:,n] + b1[n]
  for (int i = tid; i < H1_DIM * E_DIM; i += 256) {
    int n = i >> 6, k = i & 63;
    float w = W1[(64 + k) * H1_DIM + n] - W1[(128 + k) * H1_DIM + n]
            + qs[k] * W1[(192 + k) * H1_DIM + n];
    Weff[i] = (_Float16)w;
  }
  if (tid < H1_DIM) {
    float acc = b1[tid];
    for (int k = 0; k < E_DIM; ++k)
      acc += qs[k] * (W1[k * H1_DIM + tid] + W1[(128 + k) * H1_DIM + tid]);
    cvec[tid] = acc;
  }
  __syncthreads();

  // ---------------- GEMM1: [208,64] x [64,80] -> sigmoid -> h1 --------------
  // 13 M-tiles x 5 N-tiles = 65 tiles, K = 2 steps of 32.
  for (int tile = wave; tile < 13 * 5; tile += 8) {
    int mt = tile / 5, nt = tile - mt * 5;
    int rowA = mt * 16 + (lane & 15);
    int colB = nt * 16 + (lane & 15);
    int aoff = ((lane >> 4) & 1) * 16;   // A: lane-half K split 0..15 / 16..31
    int boff = ((lane >> 4) & 1) * 16;   // B: lane-half K split 0..15 / 16..31
    v8f acc = {};
#pragma unroll
    for (int ks = 0; ks < 2; ++ks) {
      const _Float16* ap = keysA + rowA * E_DIM + ks * 32 + aoff;
      const _Float16* bp = Weff  + colB * E_DIM + ks * 32 + boff;
      v16h A  = load_frag(ap, ap + 8);
      v16h Bf = load_frag(bp, bp + 8);
      acc = __builtin_amdgcn_wmma_f32_16x16x32_f16(false, A, false, Bf,
                                                   (short)0, acc, false, false);
    }
    float bias = cvec[colB];
#pragma unroll
    for (int r = 0; r < 8; ++r) {
      int mrow = mt * 16 + r + ((lane >> 4) & 1) * 8;
      float s = sigmoidf_fast(acc[r] + bias);
      h1[mrow * H1_PAD + colB] = (_Float16)s;
    }
  }
  __syncthreads();

  // ---------------- GEMM2: [208,96p] x [96p,48p] -> sigmoid -> h2 -----------
  // 13 M-tiles x 3 N-tiles = 39 tiles, K = 3 steps of 32 (zero padded).
  for (int tile = wave; tile < 13 * 3; tile += 8) {
    int mt = tile / 3, nt = tile - mt * 3;
    int rowA = mt * 16 + (lane & 15);
    int colB = nt * 16 + (lane & 15);
    int aoff = ((lane >> 4) & 1) * 16;
    int boff = ((lane >> 4) & 1) * 16;
    v8f acc = {};
#pragma unroll
    for (int ks = 0; ks < 3; ++ks) {
      const _Float16* ap = h1  + rowA * H1_PAD + ks * 32 + aoff;
      const _Float16* bp = W2s + colB * H1_PAD + ks * 32 + boff;
      v16h A  = load_frag(ap, ap + 8);
      v16h Bf = load_frag(bp, bp + 8);
      acc = __builtin_amdgcn_wmma_f32_16x16x32_f16(false, A, false, Bf,
                                                   (short)0, acc, false, false);
    }
    float bias = b2s[colB];
#pragma unroll
    for (int r = 0; r < 8; ++r) {
      int mrow = mt * 16 + r + ((lane >> 4) & 1) * 8;
      if (colB < H2_DIM) {
        float s = sigmoidf_fast(acc[r] + bias);
        h2[mrow * H2_DIM + colB] = (_Float16)s;
      }
    }
  }
  __syncthreads();

  // ---------------- GEMM3 (matvec) + mask -> scores -------------------------
  if (tid < T_DIM) {
    float acc = b3[0];
#pragma unroll 8
    for (int j = 0; j < H2_DIM; ++j)
      acc += (float)h2[tid * H2_DIM + j] * w3s[j];
    unsigned char m = key_masks[(size_t)b * T_DIM + tid];
    scores[tid] = m ? acc : NEG_PAD_F;
  }
  __syncthreads();

  // ---------------- masked softmax over T -----------------------------------
  float v = (tid < T_DIM) ? scores[tid] : -INFINITY;
  red[tid] = v;
  __syncthreads();
  for (int off = 128; off > 0; off >>= 1) {
    if (tid < off) red[tid] = fmaxf(red[tid], red[tid + off]);
    __syncthreads();
  }
  float mx = red[0];
  __syncthreads();
  float e = (tid < T_DIM) ? __expf(scores[tid] - mx) : 0.0f;
  red[tid] = e;
  __syncthreads();
  for (int off = 128; off > 0; off >>= 1) {
    if (tid < off) red[tid] += red[tid + off];
    __syncthreads();
  }
  float rden = __builtin_amdgcn_rcpf(red[0]);
  __syncthreads();
  if (tid < T_DIM) scores[tid] = e * rden;
  __syncthreads();

  // ---------------- out[b,e] = sum_t w[t] * keys[b,t,e] ---------------------
  {
    int e0   = tid & 63;
    int tsub = tid >> 6;
    float part = 0.0f;
    for (int t = tsub; t < T_DIM; t += 4)
      part += scores[t] * keysb[t * E_DIM + e0];
    outacc[tid] = part;
  }
  __syncthreads();
  if (tid < 64) {
    out[(size_t)b * E_DIM + tid] =
        outacc[tid] + outacc[64 + tid] + outacc[128 + tid] + outacc[192 + tid];
  }
}

extern "C" void kernel_launch(void* const* d_in, const int* in_sizes, int n_in,
                              void* d_out, int out_size, void* d_ws, size_t ws_size,
                              hipStream_t stream) {
  const float* queries = (const float*)d_in[0];
  const float* keys    = (const float*)d_in[1];
  const unsigned char* key_masks = (const unsigned char*)d_in[2];
  const float* W1 = (const float*)d_in[3];
  const float* b1 = (const float*)d_in[4];
  const float* W2 = (const float*)d_in[5];
  const float* b2 = (const float*)d_in[6];
  const float* W3 = (const float*)d_in[7];
  const float* b3 = (const float*)d_in[8];
  float* out = (float*)d_out;

  int B = in_sizes[0] / E_DIM;  // 4096
  attn_seq_pool_kernel<<<B, 256, SMEM_BYTES, stream>>>(
      queries, keys, key_masks, W1, b1, W2, b2, W3, b3, out);
}
